// ENNMultiheadAttention_89498528514069
// MI455X (gfx1250) — compile-verified, matrix operand order checked by reading, not run on hardware
//
#include <hip/hip_runtime.h>
#include <hip/hip_bf16.h>
#include <math.h>

// Problem constants (from the reference)
#define B_ 2
#define S_ 2048
#define D_ 1024
#define H_ 16
#define HD_ 64
#define M_ (B_ * S_)   // 4096 rows for all projections

typedef __attribute__((ext_vector_type(16))) _Float16 v16h;
typedef __attribute__((ext_vector_type(8)))  float    v8f;
typedef unsigned int u32x4 __attribute__((ext_vector_type(4)));
typedef int          i32x4 __attribute__((ext_vector_type(4)));
typedef int          i32x8 __attribute__((ext_vector_type(8)));

// Tensor Data Mover availability (device pass only; arity differs by toolchain)
#if defined(__HIP_DEVICE_COMPILE__) && __has_builtin(__builtin_amdgcn_tensor_load_to_lds)
#define USE_TDM 1
#else
#define USE_TDM 0
#endif

// ---------------------------------------------------------------------------
// WMMA wrapper: D = A(16x32 f16) x B(32x16 f16) + C(16x16 f32)
// ---------------------------------------------------------------------------
__device__ inline v8f wmma_f16(v16h a, v16h b, v8f c) {
    return __builtin_amdgcn_wmma_f32_16x16x32_f16(
        /*neg_a=*/false, a, /*neg_b=*/false, b,
        /*c_mod=*/(short)0, c, /*reuse_a=*/false, /*reuse_b=*/false);
}

// A-fragment (16x32, 16-bit) per ISA layout:
//   lane L holds row M = L%16; lanes 0-15 hold K={0..7,16..23},
//   lanes 16-31 hold K={8..15,24..31}; VGPR v holds consecutive pair (k,k+1).
// Also used as the B-fragment of a *transposed* operand (lane = column N),
// since the B layout mirrors A with M<->N.
__device__ inline v16h load_A(const _Float16* __restrict__ p, int ld) {
    const int lane = threadIdx.x & 31;
    const int m  = lane & 15;
    const int kb = (lane >> 4) << 3;   // 0 or 8
    const _Float16* row = p + (size_t)m * ld;
    v16h a;
#pragma unroll
    for (int v = 0; v < 8; ++v) {
        const int k = (v < 4) ? (kb + 2 * v) : (16 + kb + 2 * (v - 4));
        a[2 * v]     = row[k];
        a[2 * v + 1] = row[k + 1];
    }
    return a;
}

// ---------------------------------------------------------------------------
// Tensor Data Mover: async 2D tile (global -> LDS), fp16 elements.
//   width  = tile_dim0 (contiguous elements per row)
//   height = tile_dim1 (rows)
//   stride = row stride of the global matrix, in elements
// Descriptor packing per CDNA5 ISA ch.8 (D# groups 0/1; 2D -> groups 2/3 zero).
// ---------------------------------------------------------------------------
#if USE_TDM
__device__ inline void tdm_load_2d(unsigned lds_off, const void* gaddr,
                                   unsigned width, unsigned height,
                                   unsigned stride) {
    const unsigned long long ga = (unsigned long long)(uintptr_t)gaddr;
    u32x4 g0;
    g0[0] = 1u;                                   // count=1, user descriptor
    g0[1] = lds_off;                              // lds_addr (bytes)
    g0[2] = (unsigned)ga;                         // global_addr[31:0]
    g0[3] = (unsigned)((ga >> 32) & 0x1FFFFFFu)   // global_addr[56:32]
          | (2u << 30);                           // type = 2 ("image")
    i32x8 g1;
    g1[0] = 0x00010000;                           // data_size=1 (2 bytes)
    g1[1] = (int)((width & 0xFFFFu) << 16);       // tensor_dim0[15:0]
    g1[2] = (int)(((width >> 16) & 0xFFFFu)       // tensor_dim0[31:16]
          | ((height & 0xFFFFu) << 16));          // tensor_dim1[15:0]
    g1[3] = (int)(((height >> 16) & 0xFFFFu)      // tensor_dim1[31:16]
          | ((width & 0xFFFFu) << 16));           // tile_dim0
    g1[4] = (int)(height & 0xFFFFu);              // tile_dim1 (tile_dim2=0)
    g1[5] = (int)stride;                          // tensor_dim0_stride[31:0]
    g1[6] = 0;
    g1[7] = 0;
    i32x4 z4 = (i32x4)0;
#if __clang_major__ <= 22
    __builtin_amdgcn_tensor_load_to_lds(g0, g1, z4, z4, 0);
#else
    i32x8 z8 = (i32x8)0;
    __builtin_amdgcn_tensor_load_to_lds(g0, g1, z4, z4, z8, 0);
#endif
}
#endif

// ---------------------------------------------------------------------------
// fp32 -> fp16 conversion (row-major copy)
// ---------------------------------------------------------------------------
__global__ __launch_bounds__(256) void cvt_f32_to_f16(
    const float* __restrict__ in, _Float16* __restrict__ out, int n) {
    int i = blockIdx.x * 256 + threadIdx.x;
    if (i < n) out[i] = (_Float16)in[i];
}

// fp32 [K,N] -> fp16 transposed [N,K]
__global__ __launch_bounds__(256) void cvt_transpose_f32_f16(
    const float* __restrict__ in, _Float16* __restrict__ out,
    int Kdim, int Ndim) {
    int i = blockIdx.x * 256 + threadIdx.x;
    if (i < Kdim * Ndim) {
        const int k = i / Ndim;
        const int n = i % Ndim;
        out[(size_t)n * Kdim + k] = (_Float16)in[i];
    }
}

// ---------------------------------------------------------------------------
// GEMM + bias: C[M,N] = A[M,K](f16, row-major) * Wt[N,K](f16, transposed!)
//              + bias[N](f32)
// One wave computes a 16x64 strip (A-fragment reused across 4 WMMAs).
// 8 waves / block. OUT_MODE: 0 = f32 row-major, 1 = f16 row-major,
//                            2 = f16 scattered to [B,H,HD,S] (V-transpose)
// ---------------------------------------------------------------------------
template <int OUT_MODE>
__global__ __launch_bounds__(256) void gemm_bias_kernel(
    const _Float16* __restrict__ A, const _Float16* __restrict__ Wt,
    const float* __restrict__ bias, void* __restrict__ Cout,
    int M, int N, int K) {
    const int wave = threadIdx.x >> 5;
    const int gw = blockIdx.x * 8 + wave;        // global wave index
    const int tiles_n64 = N >> 6;
    const int tm = gw / tiles_n64;               // 16-row tile
    const int tn = gw % tiles_n64;               // 64-col strip

    const _Float16* Ap = A + (size_t)tm * 16 * K;
    const _Float16* Wp = Wt + (size_t)tn * 64 * K;

    v8f c[4];
#pragma unroll
    for (int t = 0; t < 4; ++t)
#pragma unroll
        for (int r = 0; r < 8; ++r) c[t][r] = 0.0f;

    for (int k = 0; k < K; k += 32) {
        if (k + 256 < K) {
            __builtin_prefetch(Ap + k + 256, 0, 0);
            __builtin_prefetch(Wp + k + 256, 0, 0);
        }
        const v16h a = load_A(Ap + k, K);
#pragma unroll
        for (int t = 0; t < 4; ++t) {
            // B-fragment of W = A-fragment of Wt rows (lane = output column)
            const v16h b = load_A(Wp + (size_t)(t * 16) * K + k, K);
            c[t] = wmma_f16(a, b, c[t]);
        }
    }

    const int lane = threadIdx.x & 31;
    const int nn = lane & 15;
    const int mbase = tm * 16 + ((lane >> 4) << 3);
#pragma unroll
    for (int t = 0; t < 4; ++t) {
        const int n = tn * 64 + t * 16 + nn;
        const float bv = bias[n];
#pragma unroll
        for (int r = 0; r < 8; ++r) {
            const int m = mbase + r;
            const float val = c[t][r] + bv;
            if (OUT_MODE == 0) {
                ((float*)Cout)[(size_t)m * N + n] = val;
            } else if (OUT_MODE == 1) {
                ((_Float16*)Cout)[(size_t)m * N + n] = (_Float16)val;
            } else {
                // (m = b*S+s, n = h*HD+d)  ->  Vt[b][h][d][s]
                const int bb = m >> 11, s = m & (S_ - 1);
                const int hh = n >> 6,  dd = n & (HD_ - 1);
                ((_Float16*)Cout)[(((size_t)bb * H_ + hh) * HD_ + dd) * S_ + s]
                    = (_Float16)val;
            }
        }
    }
}

// ---------------------------------------------------------------------------
// Flash attention: one wave per (b, h, 16-query tile).
// K-tiles (32x64) and Vt-tiles (64x32) are DMA'd into LDS by the Tensor Data
// Mover, double-buffered with s_wait_tensorcnt; P goes through LDS to turn
// the WMMA C layout into an A-fragment for the P*V WMMAs.
// ---------------------------------------------------------------------------
__global__ __launch_bounds__(32) void attn_kernel(
    const _Float16* __restrict__ Q, const _Float16* __restrict__ K,
    const _Float16* __restrict__ Vt,   // [B,H,HD,S]
    _Float16* __restrict__ Ctx) {
    const int qt = blockIdx.x;     // 0..S/16-1
    const int h  = blockIdx.y;     // 0..H-1
    const int b  = blockIdx.z;     // 0..B-1
    const int lane = threadIdx.x & 31;
    const float scale = 0.125f;    // 1/sqrt(HD), HD=64

    __shared__ _Float16 lds_k[2][32 * HD_];   // K tile: 32 keys x 64 d
    __shared__ _Float16 lds_v[2][HD_ * 32];   // Vt tile: 64 d x 32 keys
    __shared__ _Float16 lds_p[16 * 32];       // P tile, row-major

    const _Float16* Qbase = Q + ((size_t)(b * S_) + qt * 16) * D_ + h * HD_;
    const v16h qa0 = load_A(Qbase, D_);        // d = 0..31
    const v16h qa1 = load_A(Qbase + 32, D_);   // d = 32..63

    const _Float16* Kh0 = K + (size_t)(b * S_) * D_ + h * HD_;        // ld D_
    const _Float16* Vh0 = Vt + ((size_t)(b * H_) + h) * HD_ * S_;     // ld S_

    float mrow[8], lrow[8];
    v8f o[4];
#pragma unroll
    for (int r = 0; r < 8; ++r) { mrow[r] = -1e30f; lrow[r] = 0.0f; }
#pragma unroll
    for (int t = 0; t < 4; ++t)
#pragma unroll
        for (int r = 0; r < 8; ++r) o[t][r] = 0.0f;

    const int nn = lane & 15;
    const int mb = (lane >> 4) << 3;

#if USE_TDM
    tdm_load_2d((unsigned)(uintptr_t)&lds_k[0][0], Kh0, HD_, 32, D_);
    tdm_load_2d((unsigned)(uintptr_t)&lds_v[0][0], Vh0, 32, HD_, S_);
#endif
    int buf = 0;

    for (int kt = 0; kt < S_; kt += 32) {
        const _Float16* Kb;
        const _Float16* Vb;
        int ldk, ldv;
#if USE_TDM
        if (kt + 32 < S_) {
            tdm_load_2d((unsigned)(uintptr_t)&lds_k[buf ^ 1][0],
                        Kh0 + (size_t)(kt + 32) * D_, HD_, 32, D_);
            tdm_load_2d((unsigned)(uintptr_t)&lds_v[buf ^ 1][0],
                        Vh0 + (kt + 32), 32, HD_, S_);
            __builtin_amdgcn_s_wait_tensorcnt(2);  // current tile ready
        } else {
            __builtin_amdgcn_s_wait_tensorcnt(0);
        }
        Kb = &lds_k[buf][0];  ldk = HD_;
        Vb = &lds_v[buf][0];  ldv = 32;
#else
        Kb = Kh0 + (size_t)kt * D_;  ldk = D_;
        Vb = Vh0 + kt;               ldv = S_;
#endif

        // Scores S = Q*K^T for keys kt..kt+31 (two 16x16 tiles).
        v8f s0, s1;
#pragma unroll
        for (int r = 0; r < 8; ++r) { s0[r] = 0.0f; s1[r] = 0.0f; }
        s0 = wmma_f16(qa0, load_A(Kb, ldk), s0);
        s0 = wmma_f16(qa1, load_A(Kb + 32, ldk), s0);
        s1 = wmma_f16(qa0, load_A(Kb + (size_t)16 * ldk, ldk), s1);
        s1 = wmma_f16(qa1, load_A(Kb + (size_t)16 * ldk + 32, ldk), s1);

#pragma unroll
        for (int r = 0; r < 8; ++r) { s0[r] *= scale; s1[r] *= scale; }

        // Row max over 32 key columns (xor masks 1..8 stay inside the
        // 16-lane half that owns the row in the C layout).
        float pm[8];
#pragma unroll
        for (int r = 0; r < 8; ++r) pm[r] = fmaxf(s0[r], s1[r]);
#pragma unroll
        for (int mask = 1; mask <= 8; mask <<= 1)
#pragma unroll
            for (int r = 0; r < 8; ++r)
                pm[r] = fmaxf(pm[r], __shfl_xor(pm[r], mask, 32));

        float corr[8];
#pragma unroll
        for (int r = 0; r < 8; ++r) {
            const float mn = fmaxf(mrow[r], pm[r]);
            corr[r] = __expf(mrow[r] - mn);
            mrow[r] = mn;
        }

        float rs[8];
#pragma unroll
        for (int r = 0; r < 8; ++r) {
            s0[r] = __expf(s0[r] - mrow[r]);
            s1[r] = __expf(s1[r] - mrow[r]);
            rs[r] = s0[r] + s1[r];
        }
#pragma unroll
        for (int mask = 1; mask <= 8; mask <<= 1)
#pragma unroll
            for (int r = 0; r < 8; ++r)
                rs[r] += __shfl_xor(rs[r], mask, 32);

#pragma unroll
        for (int r = 0; r < 8; ++r) lrow[r] = lrow[r] * corr[r] + rs[r];
#pragma unroll
        for (int t = 0; t < 4; ++t)
#pragma unroll
            for (int r = 0; r < 8; ++r) o[t][r] *= corr[r];

        // P: C-layout -> row-major LDS -> A-fragment
#pragma unroll
        for (int r = 0; r < 8; ++r) {
            lds_p[(mb + r) * 32 + nn]      = (_Float16)s0[r];
            lds_p[(mb + r) * 32 + 16 + nn] = (_Float16)s1[r];
        }
        __syncthreads();
        const v16h pa = load_A(lds_p, 32);
        __syncthreads();

        // O += P * V; B-fragment of V = A-fragment of Vt rows (lane = d)
#pragma unroll
        for (int t = 0; t < 4; ++t) {
            const v16h vb = load_A(Vb + (size_t)(t * 16) * ldv, ldv);
            o[t] = wmma_f16(pa, vb, o[t]);
        }
        buf ^= 1;
    }

    // Normalize and store ctx (f16) at [b, qt*16 + m, h*HD + d]
    _Float16* Cb = Ctx + ((size_t)(b * S_) + qt * 16) * D_ + h * HD_;
#pragma unroll
    for (int t = 0; t < 4; ++t)
#pragma unroll
        for (int r = 0; r < 8; ++r)
            Cb[(size_t)(mb + r) * D_ + t * 16 + nn] =
                (_Float16)(o[t][r] / lrow[r]);
}

// ---------------------------------------------------------------------------
// Host orchestration
// ---------------------------------------------------------------------------
extern "C" void kernel_launch(void* const* d_in, const int* in_sizes, int n_in,
                              void* d_out, int out_size, void* d_ws, size_t ws_size,
                              hipStream_t stream) {
    (void)in_sizes; (void)n_in; (void)out_size; (void)ws_size;

    const float* x  = (const float*)d_in[0];
    const float* Wq = (const float*)d_in[1];
    const float* bq = (const float*)d_in[2];
    const float* Wk = (const float*)d_in[3];
    const float* bk = (const float*)d_in[4];
    const float* Wv = (const float*)d_in[5];
    const float* bv = (const float*)d_in[6];
    const float* Wo = (const float*)d_in[7];
    const float* bo = (const float*)d_in[8];
    float* out = (float*)d_out;

    const int NX = M_ * D_;   // 4194304
    const int NW = D_ * D_;   // 1048576

    _Float16* xh  = (_Float16*)d_ws;
    _Float16* Wqt = xh  + NX;   // transposed f16 weights [N,K]
    _Float16* Wkt = Wqt + NW;
    _Float16* Wvt = Wkt + NW;
    _Float16* Wot = Wvt + NW;
    _Float16* Qh  = Wot + NW;   // [B*S, D]
    _Float16* Kh  = Qh  + NX;   // [B*S, D]
    _Float16* Vt  = Kh  + NX;   // [B, H, HD, S]
    _Float16* Ctx = Vt  + NX;   // [B*S, D]

    // fp32 -> fp16 (x row-major; weights transposed)
    cvt_f32_to_f16<<<(NX + 255) / 256, 256, 0, stream>>>(x, xh, NX);
    cvt_transpose_f32_f16<<<(NW + 255) / 256, 256, 0, stream>>>(Wq, Wqt, D_, D_);
    cvt_transpose_f32_f16<<<(NW + 255) / 256, 256, 0, stream>>>(Wk, Wkt, D_, D_);
    cvt_transpose_f32_f16<<<(NW + 255) / 256, 256, 0, stream>>>(Wv, Wvt, D_, D_);
    cvt_transpose_f32_f16<<<(NW + 255) / 256, 256, 0, stream>>>(Wo, Wot, D_, D_);

    // Q/K/V projections: 4096 waves, 8 waves/block -> 512 blocks
    const int gemm_blocks = (M_ / 16) * (D_ / 64) / 8;
    gemm_bias_kernel<1><<<gemm_blocks, 256, 0, stream>>>(xh, Wqt, bq, Qh, M_, D_, D_);
    gemm_bias_kernel<1><<<gemm_blocks, 256, 0, stream>>>(xh, Wkt, bk, Kh, M_, D_, D_);
    gemm_bias_kernel<2><<<gemm_blocks, 256, 0, stream>>>(xh, Wvt, bv, Vt, M_, D_, D_);

    // Fused attention (WMMA + TDM double-buffered K/V staging)
    attn_kernel<<<dim3(S_ / 16, H_, B_), 32, 0, stream>>>(Qh, Kh, Vt, Ctx);

    // Output projection (WMMA, f32 out)
    gemm_bias_kernel<0><<<gemm_blocks, 256, 0, stream>>>(Ctx, Wot, bo, out, M_, D_, D_);
}
